// Attn_59064390255156
// MI455X (gfx1250) — matrix operand order repaired and checked by reading, not compiled, not run on hardware
//
#include <hip/hip_runtime.h>

// ---------------------------------------------------------------------------
// MI455X (gfx1250, wave32) fused attention:
//   1) convert x / W_qkv / W_out to bf16
//   2) QKV GEMM (bf16 WMMA, f32 acc, 32x64 per wave) fused with bias +
//      RMSNorm + RoPE epilogue -> q[B,H,L,dh], k[B,H,L,dh], v^T [B,H,dh,L]
//   3) flash attention per (b, h, q-frame); frame-aligned block-causal mask
//      => dense key frames; K/V tiles staged into LDS with
//      global_load_async_to_lds_b128 (ASYNCcnt) and consumed by WMMA
//   4) output projection GEMM (bf16 WMMA, f32 acc) + bias -> d_out f32
// ---------------------------------------------------------------------------

typedef __attribute__((ext_vector_type(16))) __bf16 v16bf;
typedef __attribute__((ext_vector_type(8)))  float  v8f;

#define B_   2
#define L_   2048
#define D_   1024
#define H_   16
#define DH_  64
#define FR_  128   // tokens per frame
#define NF_  16    // frames

__device__ inline v8f vzero8() {
  v8f z;
#pragma unroll
  for (int i = 0; i < 8; ++i) z[i] = 0.f;
  return z;
}

__device__ inline unsigned short f2bf(float f) {
  unsigned int u = __float_as_uint(f);
  u += 0x7FFFu + ((u >> 16) & 1u);   // round to nearest even
  return (unsigned short)(u >> 16);
}

union Frag { v16bf v; uint4 q[2]; };

// A fragment (16x32 bf16, MxK): lane = row (lane&15); per ISA table the 16
// elements are two contiguous 8-element K runs at k0+8*half and k0+16+8*half.
__device__ inline v16bf load_a_frag(const unsigned short* A, int lda, int m0,
                                    int k0, int lane) {
  int row  = m0 + (lane & 15);
  int half = lane >> 4;
  const unsigned short* p = A + (size_t)row * lda + k0 + 8 * half;
  Frag f;
  f.q[0] = *(const uint4*)p;          // K = k0+8h .. +7
  f.q[1] = *(const uint4*)(p + 16);   // K = k0+16+8h .. +7
  return f.v;
}

// B fragment (32x16 bf16, KxN) sourced from an NT operand (row n contiguous
// in K): lane = col n (lane&15); elements = two 8-element K runs at
// k0+8*half and k0+16+8*half (same order as A).
__device__ inline v16bf load_b_frag(const unsigned short* Bm, int ldb, int n0,
                                    int k0, int lane) {
  int nrow = n0 + (lane & 15);
  int half = lane >> 4;
  const unsigned short* p = Bm + (size_t)nrow * ldb + k0 + 8 * half;
  Frag f;
  f.q[0] = *(const uint4*)p;
  f.q[1] = *(const uint4*)(p + 16);
  return f.v;
}

__device__ inline v8f wmma_bf16(v16bf a, v16bf b, v8f c) {
  return __builtin_amdgcn_wmma_f32_16x16x32_bf16(false, a, false, b, (short)0,
                                                 c, false, false);
}

// ---------------------------------------------------------------------------
__global__ void cvt_f32_bf16(const float* __restrict__ src,
                             unsigned short* __restrict__ dst, int n) {
  int i = blockIdx.x * 256 + threadIdx.x;
  if (i < n) dst[i] = f2bf(src[i]);
}

// ---------------------------------------------------------------------------
// QKV GEMM + bias + RMSNorm + RoPE.  Block = 8 waves; each wave owns a
// 32x64 output tile (two 16-row strips sharing B fragments).
// The 64-col tile == exactly one head of one of {q,k,v}.
__global__ __launch_bounds__(256) void qkv_gemm_fused(
    const unsigned short* __restrict__ xb,    // [4096,1024] bf16
    const unsigned short* __restrict__ wq,    // [3072,1024] bf16
    const float* __restrict__ bqkv,           // [3072]
    const float* __restrict__ qg, const float* __restrict__ kg,  // [64]
    unsigned short* __restrict__ qout,        // [B,H,L,dh]
    unsigned short* __restrict__ kout,        // [B,H,L,dh]
    unsigned short* __restrict__ vtout) {     // [B,H,dh,L]
  int lane = threadIdx.x & 31, wave = threadIdx.x >> 5;
  int m0 = blockIdx.x * 256 + wave * 32;
  int n0 = blockIdx.y * 64;
  int col = lane & 15, half = lane >> 4;

  v8f acc[2][4];
#pragma unroll
  for (int s = 0; s < 2; ++s)
#pragma unroll
    for (int t = 0; t < 4; ++t) acc[s][t] = vzero8();

  for (int k0 = 0; k0 < D_; k0 += 32) {
    // prefetch the streams ~256B ahead (global_prefetch_b8)
    __builtin_prefetch(xb + (size_t)(m0 + (lane & 31)) * D_ + k0 + 128, 0, 3);
    __builtin_prefetch(wq + (size_t)(n0 + (lane & 31)) * D_ + k0 + 128, 0, 3);
    v16bf a0 = load_a_frag(xb, D_, m0, k0, lane);
    v16bf a1 = load_a_frag(xb, D_, m0 + 16, k0, lane);
#pragma unroll
    for (int t = 0; t < 4; ++t) {
      v16bf b = load_b_frag(wq, D_, n0 + 16 * t, k0, lane);
      acc[0][t] = wmma_bf16(a0, b, acc[0][t]);
      acc[1][t] = wmma_bf16(a1, b, acc[1][t]);
    }
  }

  int sec = n0 >> 10;            // 0=q 1=k 2=v
  int h   = (n0 & 1023) >> 6;    // head
  const float kC = -0.2878231366242557f;  // -ln(10000)/32

#pragma unroll
  for (int s = 0; s < 2; ++s) {
    int m0s = m0 + 16 * s;
#pragma unroll
    for (int t = 0; t < 4; ++t) {
      float bb = bqkv[n0 + 16 * t + col];
#pragma unroll
      for (int r = 0; r < 8; ++r) acc[s][t][r] += bb;
    }
    if (sec < 2) {
      const float* gam = (sec == 0) ? qg : kg;
      unsigned short* dst = (sec == 0) ? qout : kout;
      float g[4];
#pragma unroll
      for (int t = 0; t < 4; ++t) g[t] = gam[16 * t + col];
#pragma unroll
      for (int r = 0; r < 8; ++r) {
        // RMS over the 64-wide head: 4 tiles x 16 lanes of this half
        float ss = 0.f;
#pragma unroll
        for (int t = 0; t < 4; ++t) ss += acc[s][t][r] * acc[s][t][r];
        ss += __shfl_xor(ss, 1, 32);
        ss += __shfl_xor(ss, 2, 32);
        ss += __shfl_xor(ss, 4, 32);
        ss += __shfl_xor(ss, 8, 32);
        float inv = rsqrtf(ss * (1.f / 64.f) + 1e-6f);
        float e0 = acc[s][0][r] * inv * g[0];
        float e1 = acc[s][1][r] * inv * g[1];
        float e2 = acc[s][2][r] * inv * g[2];
        float e3 = acc[s][3][r] * inv * g[3];
        int m = m0s + r + 8 * half;
        float pos = (float)(m & (L_ - 1));
        float a0 = pos * __expf((float)col * kC);
        float a1 = pos * __expf((float)(16 + col) * kC);
        float c0 = __cosf(a0), s0 = __sinf(a0);
        float c1 = __cosf(a1), s1 = __sinf(a1);
        float o0 = e0 * c0 - e2 * s0;
        float o1 = e1 * c1 - e3 * s1;
        float o2 = e2 * c0 + e0 * s0;
        float o3 = e3 * c1 + e1 * s1;
        int b = m >> 11, lpos = m & (L_ - 1);
        size_t base = ((size_t)(b * H_ + h) * L_ + lpos) * DH_;
        dst[base + col]      = f2bf(o0);
        dst[base + 16 + col] = f2bf(o1);
        dst[base + 32 + col] = f2bf(o2);
        dst[base + 48 + col] = f2bf(o3);
      }
    } else {
      // v: store transposed so P@V B-fragments are contiguous loads
#pragma unroll
      for (int r = 0; r < 8; ++r) {
        int m = m0s + r + 8 * half;
        int b = m >> 11, lpos = m & (L_ - 1);
        size_t base = (size_t)(b * H_ + h) * DH_ * L_;
#pragma unroll
        for (int t = 0; t < 4; ++t)
          vtout[base + (size_t)(16 * t + col) * L_ + lpos] =
              f2bf(acc[s][t][r]);
      }
    }
  }
}

// ---------------------------------------------------------------------------
// Flash attention: one block per (b*H+h, q-frame). 8 waves x 16-row strips.
// K/V frame tiles are staged cooperatively into LDS via async-to-LDS ops.
// LDS map (bytes): [0,32768) P strips | [32768,49152) K tile | [49152,65536) V
__global__ __launch_bounds__(256) void attn_kernel(
    const unsigned short* __restrict__ q,   // [B,H,L,dh]
    const unsigned short* __restrict__ k,   // [B,H,L,dh]
    const unsigned short* __restrict__ vt,  // [B,H,dh,L]
    unsigned short* __restrict__ attnb) {   // [B,L,H*dh]
  __shared__ unsigned short sm[32768];      // 64KB
  unsigned short* Plds = sm;                // [8 waves][16][128]
  unsigned short* Klds = sm + 16384;        // [128][64]  (keys x dh)
  unsigned short* Vlds = sm + 24576;        // [64][128]  (dh x keys)

  int tid = threadIdx.x;
  int lane = tid & 31, wave = tid >> 5;
  int col = lane & 15, half = lane >> 4;
  int bh = blockIdx.x;          // 0..31
  int fq = blockIdx.y;          // 0..15
  const unsigned short* Q  = q  + (size_t)bh * L_ * DH_;
  const unsigned short* K  = k  + (size_t)bh * L_ * DH_;
  const unsigned short* VT = vt + (size_t)bh * DH_ * L_;
  unsigned short* pw = Plds + wave * 16 * FR_;

  int q0 = fq * FR_ + wave * 16;
  v16bf qa0 = load_a_frag(Q, DH_, q0, 0, lane);
  v16bf qa1 = load_a_frag(Q, DH_, q0, 32, lane);

  float m_run[8], l_run[8];
  v8f o[4];
#pragma unroll
  for (int r = 0; r < 8; ++r) { m_run[r] = -3.0e38f; l_run[r] = 0.f; }
#pragma unroll
  for (int t = 0; t < 4; ++t) o[t] = vzero8();

  int fstart = (fq == NF_ - 1) ? 1 : 0;  // last frame may not see frame 0
  for (int fk = fstart; fk <= fq; ++fk) {
    int kbase = fk * FR_;

    // ---- stage K tile (contiguous 16KB) and V tile (64 rows x 256B) ----
    __syncthreads();  // WAR: prior iteration's LDS readers must be done
    {
      unsigned long long kg = (unsigned long long)(K + (size_t)kbase * DH_);
#pragma unroll
      for (int i = 0; i < 4; ++i) {
        int chunk = i * 256 + tid;                 // 0..1023, 16B each
        unsigned loff = 32768u + (unsigned)chunk * 16u;
        unsigned long long ga = kg + (unsigned long long)chunk * 16u;
        asm volatile("global_load_async_to_lds_b128 %0, %1, off"
                     :: "v"(loff), "v"(ga) : "memory");
      }
#pragma unroll
      for (int i = 0; i < 4; ++i) {
        int chunk = i * 256 + tid;                 // 0..1023
        int d = chunk >> 4;                        // dh row 0..63
        int c = chunk & 15;                        // 16B chunk within row
        unsigned loff = 49152u + (unsigned)chunk * 16u;
        unsigned long long ga =
            (unsigned long long)(VT + (size_t)d * L_ + kbase) +
            (unsigned long long)c * 16u;
        asm volatile("global_load_async_to_lds_b128 %0, %1, off"
                     :: "v"(loff), "v"(ga) : "memory");
      }
      asm volatile("s_wait_asynccnt 0x0" ::: "memory");
    }
    __syncthreads();

    // ---- S strip: 16 x 128 = 8 tiles, K-dim = dh (2 WMMA steps) ----
    v8f s[8];
#pragma unroll
    for (int t = 0; t < 8; ++t) {
      v8f c = vzero8();
      c = wmma_bf16(qa0, load_b_frag(Klds, DH_, 16 * t, 0, lane), c);
      c = wmma_bf16(qa1, load_b_frag(Klds, DH_, 16 * t, 32, lane), c);
      s[t] = c;
    }
    // ---- online softmax, per row (r + 8*half in this half's 16 lanes) ----
#pragma unroll
    for (int r = 0; r < 8; ++r) {
      float mx = -3.0e38f;
#pragma unroll
      for (int t = 0; t < 8; ++t) {
        float v = s[t][r] * 0.125f;  // 1/sqrt(dh)
        s[t][r] = v;
        mx = fmaxf(mx, v);
      }
      mx = fmaxf(mx, __shfl_xor(mx, 1, 32));
      mx = fmaxf(mx, __shfl_xor(mx, 2, 32));
      mx = fmaxf(mx, __shfl_xor(mx, 4, 32));
      mx = fmaxf(mx, __shfl_xor(mx, 8, 32));
      float mnew = fmaxf(m_run[r], mx);
      float alpha = __expf(m_run[r] - mnew);
      m_run[r] = mnew;
      float rs = 0.f;
#pragma unroll
      for (int t = 0; t < 8; ++t) {
        float p = __expf(s[t][r] - mnew);
        s[t][r] = p;
        rs += p;
      }
      rs += __shfl_xor(rs, 1, 32);
      rs += __shfl_xor(rs, 2, 32);
      rs += __shfl_xor(rs, 4, 32);
      rs += __shfl_xor(rs, 8, 32);
      l_run[r] = l_run[r] * alpha + rs;
#pragma unroll
      for (int tt = 0; tt < 4; ++tt) o[tt][r] *= alpha;
    }
    // ---- transpose P through wave-private LDS (D-layout -> A-frag) ----
#pragma unroll
    for (int t = 0; t < 8; ++t)
#pragma unroll
      for (int r = 0; r < 8; ++r)
        pw[(r + 8 * half) * FR_ + 16 * t + col] = f2bf(s[t][r]);
    __syncthreads();  // uniform trip count across all 8 waves
    // ---- O += P @ V : K-dim = 128 keys (4 steps), 4 dh tiles ----
#pragma unroll
    for (int s4 = 0; s4 < 4; ++s4) {
      v16bf pa = load_a_frag(pw, FR_, 0, 32 * s4, lane);
#pragma unroll
      for (int tt = 0; tt < 4; ++tt) {
        v16bf vb = load_b_frag(Vlds, FR_, 16 * tt, 32 * s4, lane);
        o[tt] = wmma_bf16(pa, vb, o[tt]);
      }
    }
  }

  int b = bh >> 4, h = bh & 15;
#pragma unroll
  for (int tt = 0; tt < 4; ++tt)
#pragma unroll
    for (int r = 0; r < 8; ++r) {
      float val = o[tt][r] / l_run[r];
      int lpos = q0 + r + 8 * half;
      attnb[((size_t)(b * L_ + lpos)) * D_ + h * DH_ + 16 * tt + col] =
          f2bf(val);
    }
}

// ---------------------------------------------------------------------------
// out = attn @ W_out^T + b_out   (M=4096, N=1024, K=1024), f32 result.
// 32x64 per wave (two strips share B fragments).
__global__ __launch_bounds__(256) void out_gemm(
    const unsigned short* __restrict__ A, const unsigned short* __restrict__ W,
    const float* __restrict__ bias, float* __restrict__ C) {
  int lane = threadIdx.x & 31, wave = threadIdx.x >> 5;
  int m0 = blockIdx.x * 256 + wave * 32;
  int n0 = blockIdx.y * 64;
  int col = lane & 15, half = lane >> 4;
  v8f acc[2][4];
#pragma unroll
  for (int s = 0; s < 2; ++s)
#pragma unroll
    for (int t = 0; t < 4; ++t) acc[s][t] = vzero8();
  for (int k0 = 0; k0 < D_; k0 += 32) {
    __builtin_prefetch(A + (size_t)(m0 + (lane & 31)) * D_ + k0 + 128, 0, 3);
    __builtin_prefetch(W + (size_t)(n0 + (lane & 31)) * D_ + k0 + 128, 0, 3);
    v16bf a0 = load_a_frag(A, D_, m0, k0, lane);
    v16bf a1 = load_a_frag(A, D_, m0 + 16, k0, lane);
#pragma unroll
    for (int t = 0; t < 4; ++t) {
      v16bf b = load_b_frag(W, D_, n0 + 16 * t, k0, lane);
      acc[0][t] = wmma_bf16(a0, b, acc[0][t]);
      acc[1][t] = wmma_bf16(a1, b, acc[1][t]);
    }
  }
#pragma unroll
  for (int s = 0; s < 2; ++s)
#pragma unroll
    for (int t = 0; t < 4; ++t) {
      float bb = bias[n0 + 16 * t + col];
#pragma unroll
      for (int r = 0; r < 8; ++r)
        C[(size_t)(m0 + 16 * s + r + 8 * half) * D_ + n0 + 16 * t + col] =
            acc[s][t][r] + bb;
    }
}

// ---------------------------------------------------------------------------
extern "C" void kernel_launch(void* const* d_in, const int* in_sizes, int n_in,
                              void* d_out, int out_size, void* d_ws,
                              size_t ws_size, hipStream_t stream) {
  const float* x     = (const float*)d_in[0];  // [2,2048,1024]
  const float* Wqkv  = (const float*)d_in[1];  // [3072,1024]
  const float* bqkv  = (const float*)d_in[2];  // [3072]
  const float* Wout  = (const float*)d_in[3];  // [1024,1024]
  const float* bout  = (const float*)d_in[4];  // [1024]
  const float* qgam  = (const float*)d_in[5];  // [64]
  const float* kgam  = (const float*)d_in[6];  // [64]
  float* out = (float*)d_out;                  // [2,2048,1024]

  char* W = (char*)d_ws;  // 48 MB total
  unsigned short* xb   = (unsigned short*)(W + 0);         //  8 MB
  unsigned short* wqkb = (unsigned short*)(W + 8388608);   //  6 MB
  unsigned short* wob  = (unsigned short*)(W + 14680064);  //  2 MB
  unsigned short* qb   = (unsigned short*)(W + 16777216);  //  8 MB
  unsigned short* kb   = (unsigned short*)(W + 25165824);  //  8 MB
  unsigned short* vtb  = (unsigned short*)(W + 33554432);  //  8 MB
  unsigned short* atb  = (unsigned short*)(W + 41943040);  //  8 MB

  cvt_f32_bf16<<<(4194304 + 255) / 256, 256, 0, stream>>>(x, xb, 4194304);
  cvt_f32_bf16<<<(3145728 + 255) / 256, 256, 0, stream>>>(Wqkv, wqkb, 3145728);
  cvt_f32_bf16<<<(1048576 + 255) / 256, 256, 0, stream>>>(Wout, wob, 1048576);

  qkv_gemm_fused<<<dim3(16, 48), 256, 0, stream>>>(xb, wqkb, bqkv, qgam, kgam,
                                                   qb, kb, vtb);
  attn_kernel<<<dim3(32, 16), 256, 0, stream>>>(qb, kb, vtb, atb);
  out_gemm<<<dim3(16, 16), 256, 0, stream>>>(atb, wob, bout, out);
}
